// NMF_Nodes_89876485636351
// MI455X (gfx1250) — matrix operand order repaired and checked by reading, not compiled, not run on hardware
//
#include <hip/hip_runtime.h>
#include <hip/hip_bf16.h>

#define N_NODES 16384
#define N_FEAT  128
#define TOPIC_S 3
#define TOPIC_E 10
#define T_SL    7
#define EPSF    1e-8f

typedef __attribute__((ext_vector_type(16))) _Float16     v16h;
typedef __attribute__((ext_vector_type(8)))  _Float16     v8h;
typedef __attribute__((ext_vector_type(8)))  float        v8f;
typedef __attribute__((ext_vector_type(4)))  unsigned int u32x4;
typedef __attribute__((ext_vector_type(4)))  int          i32x4;
typedef __attribute__((ext_vector_type(8)))  int          i32x8;

union V16H { v16h v; v8h h[2]; };

// Issue one 2-D TDM tile load: global -> LDS (D# groups per ISA §8.3/8.4).
__device__ __forceinline__ void tdm_load(unsigned ldsOff, const void* gp, i32x8 g1) {
  i32x4 z4 = {};
  unsigned long long ga = (unsigned long long)gp;
  u32x4 g0 = {1u, ldsOff, (unsigned)ga,
              (((unsigned)(ga >> 32)) & 0x01FFFFFFu) | 0x80000000u}; // type=2
#if __clang_major__ >= 23
  i32x8 z8 = {};
  __builtin_amdgcn_tensor_load_to_lds(g0, g1, z4, z4, z8, 0);
#else
  __builtin_amdgcn_tensor_load_to_lds(g0, g1, z4, z4, 0);
#endif
}

// ---------------------------------------------------------------------------
// Pass 1: per-node argmax codes (7 x 4-bit packed), norms, and normalized x in
// f16 both row-major (xn) and feature-major (xt). One wave32 per node.
// ---------------------------------------------------------------------------
__global__ __launch_bounds__(256) void prep_kernel(
    const float* __restrict__ x, const float* __restrict__ H,
    _Float16* __restrict__ xn, _Float16* __restrict__ xt,
    unsigned* __restrict__ code, float* __restrict__ nrm)
{
  const int wave = threadIdx.x >> 5;
  const int lane = threadIdx.x & 31;
  const int row  = blockIdx.x * 8 + wave;

  const float* xr = x + (size_t)row * N_FEAT;
  float xv[4];
  float ss = 0.f;
  #pragma unroll
  for (int q = 0; q < 4; ++q) { xv[q] = xr[lane + 32 * q]; ss += xv[q] * xv[q]; }
  #pragma unroll
  for (int s = 1; s < 32; s <<= 1) ss += __shfl_xor(ss, s, 32);
  const float nv  = sqrtf(ss);
  const float inv = 1.0f / fmaxf(nv, EPSF);

  #pragma unroll
  for (int q = 0; q < 4; ++q) {
    const int f = lane + 32 * q;
    const _Float16 hv = (_Float16)(xv[q] * inv);
    xn[(size_t)row * N_FEAT + f] = hv;
    xt[(size_t)f * N_NODES + row] = hv;     // transposed copy for WMMA B-operand
  }
  if (lane == 0) nrm[row] = nv;

  // W[n,k,t] = sum_f x[n,f] * H[f,k,t]; H layout (F, 10, 8) => f*80 + k*8 + t
  unsigned cw = 0;
  for (int t = 0; t < T_SL; ++t) {
    float best = -3.4e38f; int bk = 0;
    for (int k = 0; k < TOPIC_E; ++k) {
      float d = 0.f;
      #pragma unroll
      for (int q = 0; q < 4; ++q) {
        const int f = lane + 32 * q;
        d += xv[q] * H[(size_t)f * (TOPIC_E * 8) + k * 8 + t];
      }
      #pragma unroll
      for (int s = 1; s < 32; s <<= 1) d += __shfl_xor(d, s, 32);
      if (d > best) { best = d; bk = k; }   // strict '>' == argmax-first tie rule
    }
    cw |= (unsigned)bk << (4 * t);
  }
  if (lane == 0) code[row] = cw;
}

// ---------------------------------------------------------------------------
// Pass 2: fused  cos -> match -> w -> {rowsum, w @ x}.
// 8 waves/block; each wave owns a 16-row i-strip (128 rows per block).
// j loop in tiles of 32; tiles double-buffered in LDS, filled by the TDM so
// the DMA of tile j+32 overlaps the 16 WMMAs of tile j.
// ---------------------------------------------------------------------------
__global__ __launch_bounds__(256) void fuse_kernel(
    const float* __restrict__ x,
    const _Float16* __restrict__ xn, const _Float16* __restrict__ xt,
    const unsigned* __restrict__ code, const float* __restrict__ nrm,
    float* __restrict__ out)
{
  __shared__ __align__(32) _Float16 ldsXn[2][32 * N_FEAT];  // j-tile, row-major  (16KB)
  __shared__ __align__(32) _Float16 ldsXt[2][N_FEAT * 32];  // j-tile, feat-major (16KB)
  __shared__ __align__(32) _Float16 wbuf[8][16][32];        // per-wave w' tiles  (8KB)

  const int tid  = threadIdx.x;
  const int wave = tid >> 5, lane = tid & 31;
  const int half = (lane < 16) ? 0 : 1;
  const int ln   = lane & 15;
  const int rowBase = blockIdx.x * 128 + wave * 16;

  // A fragments of xn_i, contiguous K layout: lanes<16 K{0..15}, lanes>=16 K{16..31}
  V16H aX[4];
  {
    const _Float16* xr = xn + (size_t)(rowBase + ln) * N_FEAT;
    const int off = half ? 16 : 0;
    #pragma unroll
    for (int kc = 0; kc < 4; ++kc) {
      aX[kc].h[0] = *(const v8h*)(xr + kc * 32 + off);
      aX[kc].h[1] = *(const v8h*)(xr + kc * 32 + 8 + off);
    }
  }
  unsigned ci[8];
  #pragma unroll
  for (int r = 0; r < 8; ++r) ci[r] = code[rowBase + r + half * 8];

  v8f acc[8];
  #pragma unroll
  for (int fc = 0; fc < 8; ++fc) acc[fc] = (v8f){};
  float rs[8] = {0.f, 0.f, 0.f, 0.f, 0.f, 0.f, 0.f, 0.f};

  // Constant halves of the TDM descriptors (D# group 1), §8.4 bit layout.
  const i32x8 g1n = {0x00010000, (int)(128u << 16), (int)0x40000000u, (int)(128u << 16),
                     32, 128, 0, 0};            // xn: dim0=128 str0=128 dim1=N tile 128x32
  const i32x8 g1t = {0x00010000, (int)0x40000000u, (int)(128u << 16), (int)(32u << 16),
                     128, 16384, 0, 0};         // xt: dim0=N  str0=N   dim1=128 tile 32x128
  const unsigned ldsXnOff0 = (unsigned)(unsigned long long)(void*)&ldsXn[0][0];
  const unsigned ldsXnOff1 = (unsigned)(unsigned long long)(void*)&ldsXn[1][0];
  const unsigned ldsXtOff0 = (unsigned)(unsigned long long)(void*)&ldsXt[0][0];
  const unsigned ldsXtOff1 = (unsigned)(unsigned long long)(void*)&ldsXt[1][0];

  // Prime buffer 0 with the first j-tile.
  if (wave == 0) {
    tdm_load(ldsXnOff0, xn, g1n);
    tdm_load(ldsXtOff0, xt, g1t);
    __builtin_amdgcn_s_wait_tensorcnt(0);
  }
  __syncthreads();

  const float inv7 = 1.0f / 7.0f;
  int cur = 0;

  for (int jb = 0; jb < N_NODES; jb += 32, cur ^= 1) {
    // Prefetch next tile into the other buffer; overlaps with compute below.
    if (wave == 0 && jb + 32 < N_NODES) {
      tdm_load(cur ? ldsXnOff0 : ldsXnOff1, xn + (size_t)(jb + 32) * N_FEAT, g1n);
      tdm_load(cur ? ldsXtOff0 : ldsXtOff1, xt + (jb + 32), g1t);
    }

    const unsigned cjA = code[jb + ln], cjB = code[jb + 16 + ln];
    const float    njA = nrm[jb + ln],  njB = nrm[jb + 16 + ln];
    const _Float16* tXn = &ldsXn[cur][0];
    const _Float16* tXt = &ldsXt[cur][0];

    #pragma unroll
    for (int jj = 0; jj < 32; jj += 16) {
      // cos tile: C = Xn_i (16x128) * Xn_j^T, 4 x v_wmma_f32_16x16x32_f16
      v8f c = {};
      #pragma unroll
      for (int kc = 0; kc < 4; ++kc) {
        const v16h b = *(const v16h*)(tXn + (jj + ln) * N_FEAT + kc * 32 + half * 16);
        c = __builtin_amdgcn_wmma_f32_16x16x32_f16(false, aX[kc].v, false, b,
                                                   (short)0, c, false, false);
      }
      const unsigned cj = jj ? cjB : cjA;
      const float    nj = jj ? njB : njA;
      const int      gj = jb + jj + ln;
      #pragma unroll
      for (int r = 0; r < 8; ++r) {
        const int m = r + half * 8;
        // nibble-equality count over 7 packed argmax codes (SWAR zero-nibble)
        const unsigned xr = ci[r] ^ cj;
        const unsigned nz = (((xr & 0x77777777u) + 0x77777777u) | xr) & 0x08888888u;
        const float mt = (float)(7 - __popc(nz)) * inv7;
        float w = c[r] * mt;
        if (rowBase + m == gj) w = 1.0f;       // diagonal: w = 1
        rs[r] += w;
        wbuf[wave][m][jj + ln] = (_Float16)(w * nj);  // fold norm_j: second = w' @ xn
      }
    }
    asm volatile("s_wait_dscnt 0x0" ::: "memory");    // cross-lane LDS RAW in-wave

    // w' reloaded in A layout (16x32, K = local j)
    V16H aW;
    {
      const int off = half ? 16 : 0;
      aW.h[0] = *(const v8h*)(&wbuf[wave][ln][off]);
      aW.h[1] = *(const v8h*)(&wbuf[wave][ln][8 + off]);
    }
    #pragma unroll
    for (int fc = 0; fc < 8; ++fc) {
      const v16h b2 = *(const v16h*)(tXt + (fc * 16 + ln) * 32 + half * 16);
      acc[fc] = __builtin_amdgcn_wmma_f32_16x16x32_f16(false, aW.v, false, b2,
                                                       (short)0, acc[fc], false, false);
    }

    // Drain the prefetch DMA, then one barrier to swap buffers.
    if (wave == 0) __builtin_amdgcn_s_wait_tensorcnt(0);
    __syncthreads();
  }

  // rowsum reduction across the 16 column-lanes of each half
  #pragma unroll
  for (int r = 0; r < 8; ++r)
    #pragma unroll
    for (int s = 1; s < 16; s <<= 1) rs[r] += __shfl_xor(rs[r], s, 32);

  const float invN = 1.0f / (float)N_NODES;
  #pragma unroll
  for (int r = 0; r < 8; ++r) {
    const int gi = rowBase + r + half * 8;
    const float rv = rs[r] * invN;
    #pragma unroll
    for (int q = 0; q < 8; ++q) {
      const int f = ln + 16 * q;
      out[(size_t)gi * 256 + f]       = x[(size_t)gi * 128 + f] * rv;  // first
      out[(size_t)gi * 256 + 128 + f] = acc[q][r] * invN;              // second
    }
  }
}

// ---------------------------------------------------------------------------
extern "C" void kernel_launch(void* const* d_in, const int* in_sizes, int n_in,
                              void* d_out, int out_size, void* d_ws, size_t ws_size,
                              hipStream_t stream) {
  (void)in_sizes; (void)n_in; (void)out_size; (void)ws_size;
  const float* x = (const float*)d_in[0];
  const float* H = (const float*)d_in[1];
  float* out = (float*)d_out;

  char* ws = (char*)d_ws;
  _Float16* xn  = (_Float16*)(ws);                           // 4 MB
  _Float16* xt  = (_Float16*)(ws + (size_t)4 * 1024 * 1024); // 4 MB
  unsigned* cod = (unsigned*)(ws + (size_t)8 * 1024 * 1024); // 64 KB
  float*    nrm = (float*)   (ws + (size_t)8 * 1024 * 1024 + 64 * 1024); // 64 KB

  prep_kernel<<<N_NODES / 8, 256, 0, stream>>>(x, H, xn, xt, cod, nrm);
  fuse_kernel<<<N_NODES / 128, 256, 0, stream>>>(x, xn, xt, cod, nrm, out);
}
